// ASR_52106543235215
// MI455X (gfx1250) — matrix operand order repaired and checked by reading, not compiled, not run on hardware
//
#include <hip/hip_runtime.h>
#include <math.h>
#include <stdint.h>

// ---------------- problem sizes ----------------
#define BS   8
#define TS   2048
#define NH   4
#define DIM  512
#define VDIM 512
#define QDIM 1024
#define KS   100
#define KN   100
#define NHD  (NH * DIM)      // 2048

// ---------------- GEMM tiling ----------------
#define TILE_M   128                 // t-rows per workgroup
#define TILE_N   64                  // staged B rows (output cols) per chunk
#define NCHUNKS  (DIM / TILE_N)      // 8
#define KSTEPS   (VDIM / 32)         // 16
#define NTCHUNK  (TS / TILE_M)       // 16
// LDS rows padded by TDM: 512 bf16 = 1024B data + 16B pad -> 1040B stride
#define LDS_ROW_ELEMS 520
#define LDS_ROW_BYTES 1040
#define LDS_A_BYTES (TILE_M * LDS_ROW_BYTES)   // 133120
#define LDS_B_BYTES (TILE_N * LDS_ROW_BYTES)   // 66560
#define SMEM_GEMM   (LDS_A_BYTES + LDS_B_BYTES) // 199680

typedef __bf16 v16bf __attribute__((ext_vector_type(16)));
typedef float  v8f   __attribute__((ext_vector_type(8)));
typedef unsigned int v4u __attribute__((ext_vector_type(4)));
typedef int    v8i   __attribute__((ext_vector_type(8)));
typedef int    v4i   __attribute__((ext_vector_type(4)));

// ---------------- TDM: 2D bf16 tile -> LDS, row pad 4 dwords / 256 dwords ----
__device__ __forceinline__ void tdm_load_2d_bf16(unsigned lds_off,
                                                 const void* gptr,
                                                 unsigned tile_w,   // elems / row
                                                 unsigned tile_h,   // rows
                                                 unsigned stride_elems) {
  unsigned long long ga = (unsigned long long)gptr;
  v4u g0;
  g0.x = 1u;                                            // count=1 (valid)
  g0.y = lds_off;                                       // lds_addr (bytes)
  g0.z = (unsigned)ga;                                  // global_addr[31:0]
  g0.w = (unsigned)((ga >> 32) & 0x01FFFFFFu) | (2u << 30); // addr[56:32], type=2
  // group1: data_size=1 (2B); pad_enable, pad_interval=7 (256 dw), pad_amount=3 (4 dw)
  unsigned d0 = (1u << 16) | (1u << 20) | (7u << 22) | (3u << 25);
  v8i g1;
  g1[0] = (int)d0;
  g1[1] = (int)((tile_w & 0xFFFFu) << 16);                                  // tensor_dim0 lo
  g1[2] = (int)(((tile_w >> 16) & 0xFFFFu) | ((tile_h & 0xFFFFu) << 16));   // td0 hi | td1 lo
  g1[3] = (int)(((tile_h >> 16) & 0xFFFFu) | ((tile_w & 0xFFFFu) << 16));   // td1 hi | tile_dim0
  g1[4] = (int)(tile_h & 0xFFFFu);                                          // tile_dim1 (tile_dim2=0)
  g1[5] = (int)stride_elems;                                                // tensor_dim0_stride lo
  g1[6] = 0;
  g1[7] = 0;
  v4i z4 = {0, 0, 0, 0};
  v8i z8 = {0, 0, 0, 0, 0, 0, 0, 0};
  // 6-arg (clang-23 / therock-10.0) form: groups 2/3 unused (2-D tensor)
  __builtin_amdgcn_tensor_load_to_lds(g0, g1, z4, z4, z8, 0);
}

// ---------------- WMMA helpers (layouts per CDNA5 ISA 7.12.2) ----------------
union FragBf { v16bf v; uint4 q[2]; };

// A: 16x32 bf16. lane m=L&15 (row), g=L>>4; elems e<8 -> K=8g+e, e>=8 -> K=16+8g+(e-8)
__device__ __forceinline__ v16bf load_frag_a(const __bf16* lds, int row16, int lane, int k0) {
  FragBf f;
  const __bf16* p = lds + (size_t)(row16 + (lane & 15)) * LDS_ROW_ELEMS + k0 + 8 * (lane >> 4);
  f.q[0] = *(const uint4*)p;
  f.q[1] = *(const uint4*)(p + 16);
  return f.v;
}
// B: 32x16 bf16 from row-major [N][K] rows. lane n=L&15 (col), g=L>>4; elems e<8 -> K=8g+e, e>=8 -> K=16+8g+(e-8)
__device__ __forceinline__ v16bf load_frag_b(const __bf16* lds, int col16, int lane, int k0) {
  FragBf f;
  const __bf16* p = lds + (size_t)(col16 + (lane & 15)) * LDS_ROW_ELEMS + k0 + 8 * (lane >> 4);
  f.q[0] = *(const uint4*)p;
  f.q[1] = *(const uint4*)(p + 16);
  return f.v;
}
__device__ __forceinline__ v8f wmma_bf16(v16bf a, v16bf b, v8f c) {
  return __builtin_amdgcn_wmma_f32_16x16x32_bf16(false, a, false, b, (short)0, c, false, false);
}
__device__ __forceinline__ v8f zero8() {
  v8f z = {0.f, 0.f, 0.f, 0.f, 0.f, 0.f, 0.f, 0.f};
  return z;
}

// ---------------- small utility kernels ----------------
__global__ __launch_bounds__(256) void cvt_bf16_kernel(const float* __restrict__ in,
                                                       unsigned short* __restrict__ out,
                                                       long long n) {
  long long i = (long long)blockIdx.x * blockDim.x + threadIdx.x;
  long long stride = (long long)gridDim.x * blockDim.x;
  for (; i < n; i += stride) {
    unsigned x = __float_as_uint(in[i]);
    unsigned r = x + 0x7FFFu + ((x >> 16) & 1u);   // RNE to bf16
    out[i] = (unsigned short)(r >> 16);
  }
}

// query[b, j] = tanh(dec_state[b,:] . Wq[j,:] + bq[j]),  j = h*DIM + d
__global__ __launch_bounds__(256) void query_kernel(const float* __restrict__ dec_state,
                                                    const float* __restrict__ Wq,
                                                    const float* __restrict__ bq,
                                                    float* __restrict__ query) {
  __shared__ float ds[QDIM];
  int b = blockIdx.x;
  for (int k = threadIdx.x; k < QDIM; k += 256) ds[k] = dec_state[(size_t)b * QDIM + k];
  __syncthreads();
  int j = blockIdx.y * 256 + threadIdx.x;
  const float* row = Wq + (size_t)j * QDIM;
  float s = bq[j];
  for (int k = 0; k < QDIM; ++k) s += ds[k] * row[k];
  query[(size_t)b * NHD + j] = tanhf(s);
}

// S[k][j] = prefix_{d<j} sum_h loc_conv_w[k,h,d];  S has KN rows of (2*KS+2)
__global__ __launch_bounds__(128) void conv_prefix_kernel(const float* __restrict__ w,
                                                          float* __restrict__ S) {
  int k = threadIdx.x;
  if (k >= KN) return;
  float acc = 0.f;
  S[k * (2 * KS + 2) + 0] = 0.f;
  for (int j = 0; j < 2 * KS + 1; ++j) {
    float s = 0.f;
    for (int h = 0; h < NH; ++h) s += w[((size_t)k * NH + h) * (2 * KS + 1) + j];
    acc += s;
    S[k * (2 * KS + 2) + j + 1] = acc;
  }
}

__global__ __launch_bounds__(256) void transpose_lp_kernel(const float* __restrict__ lp,
                                                           float* __restrict__ lpT) {
  int i = blockIdx.x * 256 + threadIdx.x;          // DIM*KN = 51200
  if (i < DIM * KN) {
    int d = i / KN, k = i % KN;
    lpT[k * DIM + d] = lp[i];
  }
}

// loc[b,t,d] = tanh( sum_k conv[b,k,t] * lpT[k,d] ),
// conv[b,k,t] = inv_len * (S[k,hi] - S[k,lo]) (constant prev_att collapsed)
__global__ __launch_bounds__(256) void loc_kernel(const float* __restrict__ S,
                                                  const float* __restrict__ lpT,
                                                  const int* __restrict__ enc_len,
                                                  float* __restrict__ loc) {
  __shared__ float conv_s[TILE_M][KN];
  int tc = blockIdx.x, b = blockIdx.y;
  int t0 = tc * TILE_M;
  int len = enc_len[b];
  float inv_len = 1.0f / (float)len;
  for (int i = threadIdx.x; i < TILE_M * KN; i += 256) {
    int tl = i / KN, k = i % KN;
    int t = t0 + tl;
    int lo = KS - t; if (lo < 0) lo = 0;
    int hi = KS - t + len; if (hi > 2 * KS + 1) hi = 2 * KS + 1;
    float c = 0.f;
    if (hi > lo) c = inv_len * (S[k * (2 * KS + 2) + hi] - S[k * (2 * KS + 2) + lo]);
    conv_s[tl][k] = c;
  }
  __syncthreads();
  int d = threadIdx.x;
  for (int half = 0; half < 2; ++half, d += 256) {
    for (int tlc = 0; tlc < TILE_M / 16; ++tlc) {
      float acc[16];
#pragma unroll
      for (int i = 0; i < 16; ++i) acc[i] = 0.f;
      for (int k = 0; k < KN; ++k) {
        float lpv = lpT[k * DIM + d];
#pragma unroll
        for (int tt = 0; tt < 16; ++tt) acc[tt] += conv_s[tlc * 16 + tt][k] * lpv;
      }
      for (int tt = 0; tt < 16; ++tt)
        loc[((size_t)b * TS + t0 + tlc * 16 + tt) * DIM + d] = tanhf(acc[tt]);
    }
  }
}

// --------- fused key-GEMM + energy reduction (WMMA bf16, TDM-staged LDS) -----
// energy[b,h,t] = sum_d tanh( tanh(enc@Wk_h^T + bk) + q + loc ) * gen_w[d]
__global__ __launch_bounds__(256) void energy_kernel(const unsigned short* __restrict__ encb,
                                                     const unsigned short* __restrict__ Wkb,
                                                     const float* __restrict__ query,
                                                     const float* __restrict__ loc,
                                                     const float* __restrict__ bk,
                                                     const float* __restrict__ gen_w,
                                                     float* __restrict__ energy) {
  extern __shared__ char smem[];
  __bf16* ldsA = (__bf16*)smem;
  __bf16* ldsB = (__bf16*)(smem + LDS_A_BYTES);
  unsigned lds_base = (unsigned)(uintptr_t)smem;   // flat LDS addr low 32 = LDS offset
  const int tid = threadIdx.x, lane = tid & 31, w = tid >> 5;
  const int tc = blockIdx.x, h = blockIdx.y, b = blockIdx.z;
  const int t0 = tc * TILE_M;
  const int bh = b * NH + h;
  const int nlo = lane & 15, g = lane >> 4;

  float eacc[8];
#pragma unroll
  for (int r = 0; r < 8; ++r) eacc[r] = 0.f;

  if (tid < 32)   // A strip, loaded once per workgroup (TDM, padded rows)
    tdm_load_2d_bf16(lds_base, encb + ((size_t)b * TS + t0) * VDIM, VDIM, TILE_M, VDIM);

  for (int nc = 0; nc < NCHUNKS; ++nc) {
    __syncthreads();                                 // waves done with prior B
    if (tid < 32) {
      tdm_load_2d_bf16(lds_base + LDS_A_BYTES,
                       Wkb + ((size_t)(h * DIM + nc * TILE_N)) * VDIM, VDIM, TILE_N, VDIM);
      __builtin_amdgcn_s_wait_tensorcnt(0);          // covers A too (in-order)
    }
    __syncthreads();

    v8f c[4];
#pragma unroll
    for (int j = 0; j < 4; ++j) c[j] = zero8();
    for (int ks = 0; ks < KSTEPS; ++ks) {
      v16bf a = load_frag_a(ldsA, w * 16, lane, ks * 32);
#pragma unroll
      for (int j = 0; j < 4; ++j)
        c[j] = wmma_bf16(a, load_frag_b(ldsB, j * 16, lane, ks * 32), c[j]);
    }
    // epilogue: fold this N-chunk into per-row energy partials
#pragma unroll
    for (int j = 0; j < 4; ++j) {
      int n = nc * TILE_N + j * 16 + nlo;
      float qv = query[(size_t)b * NHD + h * DIM + n];
      float bkv = bk[h * DIM + n];
      float gw = gen_w[n];
      v8f cj = c[j];
#pragma unroll
      for (int r = 0; r < 8; ++r) {
        int t = t0 + 16 * w + 8 * g + r;
        float lv = loc[((size_t)b * TS + t) * DIM + n];
        eacc[r] += tanhf(tanhf(cj[r] + bkv) + qv + lv) * gw;
      }
    }
  }
  // reduce over the 16 columns held by each half-wave, then store
#pragma unroll
  for (int r = 0; r < 8; ++r)
    for (int off = 1; off < 16; off <<= 1) eacc[r] += __shfl_xor(eacc[r], off, 32);
  if (nlo == 0) {
#pragma unroll
    for (int r = 0; r < 8; ++r)
      energy[(size_t)bh * TS + t0 + 16 * w + 8 * g + r] = eacc[r];
  }
}

// softmax over t (masked by enc_len), temperature 0.5 -> *2
__global__ __launch_bounds__(256) void softmax_kernel(const float* __restrict__ energy,
                                                      const int* __restrict__ enc_len,
                                                      const float* __restrict__ gen_b,
                                                      float* __restrict__ attn) {
  __shared__ float red[256];
  int bh = blockIdx.x;
  int len = enc_len[bh >> 2];
  float gb = gen_b[0];
  const float* e = energy + (size_t)bh * TS;
  float m = -3.0e38f;
  for (int t = threadIdx.x; t < TS; t += 256)
    if (t < len) m = fmaxf(m, (e[t] + gb) * 2.0f);
  red[threadIdx.x] = m; __syncthreads();
  for (int s = 128; s > 0; s >>= 1) {
    if (threadIdx.x < s) red[threadIdx.x] = fmaxf(red[threadIdx.x], red[threadIdx.x + s]);
    __syncthreads();
  }
  m = red[0]; __syncthreads();
  float sum = 0.f;
  for (int t = threadIdx.x; t < TS; t += 256)
    if (t < len) sum += expf((e[t] + gb) * 2.0f - m);
  red[threadIdx.x] = sum; __syncthreads();
  for (int s = 128; s > 0; s >>= 1) {
    if (threadIdx.x < s) red[threadIdx.x] += red[threadIdx.x + s];
    __syncthreads();
  }
  float inv = 1.0f / red[0];
  for (int t = threadIdx.x; t < TS; t += 256) {
    float v = (t < len) ? expf((e[t] + gb) * 2.0f - m) * inv : 0.f;
    attn[(size_t)bh * TS + t] = v;
  }
}

// --------- fused value-GEMM + attn-weighted reduction over t (WMMA) ----------
// ctx_part[bh, tc, v] = sum_{t in tile} attn[bh,t] * tanh(enc@Wv_h^T + bv)[t,v]
__global__ __launch_bounds__(256) void value_ctx_kernel(const unsigned short* __restrict__ encb,
                                                        const unsigned short* __restrict__ Wvb,
                                                        const float* __restrict__ attn,
                                                        const float* __restrict__ bv,
                                                        float* __restrict__ ctx_part) {
  extern __shared__ char smem[];
  __bf16* ldsA = (__bf16*)smem;
  __bf16* ldsB = (__bf16*)(smem + LDS_A_BYTES);
  float* ldsCtx = (float*)(smem + SMEM_GEMM);       // [8 waves][TILE_N]
  unsigned lds_base = (unsigned)(uintptr_t)smem;
  const int tid = threadIdx.x, lane = tid & 31, w = tid >> 5;
  const int tc = blockIdx.x, h = blockIdx.y, b = blockIdx.z;
  const int t0 = tc * TILE_M;
  const int bh = b * NH + h;
  const int nlo = lane & 15, g = lane >> 4;

  float attnv[8];
#pragma unroll
  for (int r = 0; r < 8; ++r)
    attnv[r] = attn[(size_t)bh * TS + t0 + 16 * w + 8 * g + r];

  if (tid < 32)
    tdm_load_2d_bf16(lds_base, encb + ((size_t)b * TS + t0) * VDIM, VDIM, TILE_M, VDIM);

  for (int nc = 0; nc < NCHUNKS; ++nc) {
    __syncthreads();
    if (tid < 32) {
      tdm_load_2d_bf16(lds_base + LDS_A_BYTES,
                       Wvb + ((size_t)(h * VDIM + nc * TILE_N)) * VDIM, VDIM, TILE_N, VDIM);
      __builtin_amdgcn_s_wait_tensorcnt(0);
    }
    __syncthreads();

    v8f c[4];
#pragma unroll
    for (int j = 0; j < 4; ++j) c[j] = zero8();
    for (int ks = 0; ks < KSTEPS; ++ks) {
      v16bf a = load_frag_a(ldsA, w * 16, lane, ks * 32);
#pragma unroll
      for (int j = 0; j < 4; ++j)
        c[j] = wmma_bf16(a, load_frag_b(ldsB, j * 16, lane, ks * 32), c[j]);
    }
#pragma unroll
    for (int j = 0; j < 4; ++j) {
      int n = nc * TILE_N + j * 16 + nlo;
      float bvv = bv[h * VDIM + n];
      v8f cj = c[j];
      float cacc = 0.f;
#pragma unroll
      for (int r = 0; r < 8; ++r) cacc += attnv[r] * tanhf(cj[r] + bvv);
      cacc += __shfl_xor(cacc, 16, 32);             // combine row halves (same n)
      if (g == 0) ldsCtx[w * TILE_N + j * 16 + nlo] = cacc;
    }
    __syncthreads();
    if (tid < TILE_N) {
      float s = 0.f;
#pragma unroll
      for (int ww = 0; ww < 8; ++ww) s += ldsCtx[ww * TILE_N + tid];
      ctx_part[((size_t)bh * NTCHUNK + tc) * VDIM + nc * TILE_N + tid] = s;
    }
  }
}

// ctx[b, h*VDIM + v] = sum_tc ctx_part[bh, tc, v]
__global__ __launch_bounds__(256) void ctx_reduce_kernel(const float* __restrict__ part,
                                                         float* __restrict__ ctx) {
  int i = blockIdx.x * 256 + threadIdx.x;          // BS*NHD = 16384
  if (i >= BS * NHD) return;
  int b = i / NHD, u = i % NHD;
  int h = u / VDIM, v = u % VDIM;
  int bh = b * NH + h;
  float s = 0.f;
  for (int tcc = 0; tcc < NTCHUNK; ++tcc)
    s += part[((size_t)bh * NTCHUNK + tcc) * VDIM + v];
  ctx[i] = s;
}

// context[b, v] = ctx[b,:] . merge_w[v,:] + merge_b[v]
__global__ __launch_bounds__(256) void merge_kernel(const float* __restrict__ ctx,
                                                    const float* __restrict__ mw,
                                                    const float* __restrict__ mb,
                                                    float* __restrict__ out) {
  __shared__ float cs[NHD];
  int b = blockIdx.x;
  for (int i = threadIdx.x; i < NHD; i += 256) cs[i] = ctx[(size_t)b * NHD + i];
  __syncthreads();
  for (int v = threadIdx.x; v < VDIM; v += 256) {
    const float* row = mw + (size_t)v * NHD;
    float s = mb[v];
    for (int u = 0; u < NHD; ++u) s += cs[u] * row[u];
    out[(size_t)b * VDIM + v] = s;
  }
}

// ---------------- workspace layout (bytes, 256-aligned) ----------------
static constexpr size_t OFF_ENC    = 0;                                   // bf16 enc_feat
static constexpr size_t SZ_ENC     = (size_t)BS * TS * VDIM * 2;          // 16 MB
static constexpr size_t OFF_WK     = OFF_ENC + SZ_ENC;
static constexpr size_t SZ_W       = (size_t)NHD * VDIM * 2;              // 2 MB
static constexpr size_t OFF_WV     = OFF_WK + SZ_W;
static constexpr size_t OFF_QUERY  = OFF_WV + SZ_W;
static constexpr size_t SZ_QUERY   = (size_t)BS * NHD * 4;
static constexpr size_t OFF_S      = OFF_QUERY + SZ_QUERY;
static constexpr size_t OFF_LPT    = OFF_S + 81920;                       // S: 80800 B
static constexpr size_t OFF_LOC    = OFF_LPT + 204800;                    // lpT: 200 KB
static constexpr size_t SZ_LOC     = (size_t)BS * TS * DIM * 4;           // 32 MB
static constexpr size_t OFF_ENERGY = OFF_LOC + SZ_LOC;
static constexpr size_t SZ_ENERGY  = (size_t)BS * NH * TS * 4;
static constexpr size_t OFF_CTXP   = OFF_ENERGY + SZ_ENERGY;
static constexpr size_t SZ_CTXP    = (size_t)BS * NH * NTCHUNK * VDIM * 4;
static constexpr size_t OFF_CTX    = OFF_CTXP + SZ_CTXP;

extern "C" void kernel_launch(void* const* d_in, const int* in_sizes, int n_in,
                              void* d_out, int out_size, void* d_ws, size_t ws_size,
                              hipStream_t stream) {
  (void)in_sizes; (void)n_in; (void)out_size; (void)ws_size;
  const float* dec_state  = (const float*)d_in[0];
  const float* enc_feat   = (const float*)d_in[1];
  const int*   enc_len    = (const int*)d_in[2];
  const float* Wq         = (const float*)d_in[3];
  const float* bq         = (const float*)d_in[4];
  const float* Wk         = (const float*)d_in[5];
  const float* bk         = (const float*)d_in[6];
  const float* Wv         = (const float*)d_in[7];
  const float* bv         = (const float*)d_in[8];
  const float* loc_conv_w = (const float*)d_in[9];
  const float* loc_proj_w = (const float*)d_in[10];
  const float* gen_w      = (const float*)d_in[11];
  const float* gen_b      = (const float*)d_in[12];
  const float* merge_w    = (const float*)d_in[13];
  const float* merge_b    = (const float*)d_in[14];

  char* ws = (char*)d_ws;
  unsigned short* enc_bf = (unsigned short*)(ws + OFF_ENC);
  unsigned short* wk_bf  = (unsigned short*)(ws + OFF_WK);
  unsigned short* wv_bf  = (unsigned short*)(ws + OFF_WV);
  float* queryb  = (float*)(ws + OFF_QUERY);
  float* Sbuf    = (float*)(ws + OFF_S);
  float* lpT     = (float*)(ws + OFF_LPT);
  float* locbuf  = (float*)(ws + OFF_LOC);
  float* energyb = (float*)(ws + OFF_ENERGY);
  float* ctxp    = (float*)(ws + OFF_CTXP);
  float* ctxb    = (float*)(ws + OFF_CTX);

  float* attn_out = (float*)d_out;                       // [BS, NH, TS]
  float* ctx_out  = attn_out + (size_t)BS * NH * TS;     // [BS, VDIM]

  cvt_bf16_kernel<<<dim3(2048), 256, 0, stream>>>(enc_feat, enc_bf, (long long)BS * TS * VDIM);
  cvt_bf16_kernel<<<dim3(512), 256, 0, stream>>>(Wk, wk_bf, (long long)NHD * VDIM);
  cvt_bf16_kernel<<<dim3(512), 256, 0, stream>>>(Wv, wv_bf, (long long)NHD * VDIM);

  query_kernel<<<dim3(BS, NHD / 256), 256, 0, stream>>>(dec_state, Wq, bq, queryb);
  conv_prefix_kernel<<<1, 128, 0, stream>>>(loc_conv_w, Sbuf);
  transpose_lp_kernel<<<dim3((DIM * KN + 255) / 256), 256, 0, stream>>>(loc_proj_w, lpT);
  loc_kernel<<<dim3(NTCHUNK, BS), 256, 0, stream>>>(Sbuf, lpT, enc_len, locbuf);

  energy_kernel<<<dim3(NTCHUNK, NH, BS), 256, SMEM_GEMM, stream>>>(
      enc_bf, wk_bf, queryb, locbuf, bk, gen_w, energyb);

  softmax_kernel<<<dim3(BS * NH), 256, 0, stream>>>(energyb, enc_len, gen_b, attn_out);

  value_ctx_kernel<<<dim3(NTCHUNK, NH, BS), 256, SMEM_GEMM + 8 * TILE_N * 4, stream>>>(
      enc_bf, wv_bf, attn_out, bv, ctxp);

  ctx_reduce_kernel<<<dim3((BS * NHD + 255) / 256), 256, 0, stream>>>(ctxp, ctxb);
  merge_kernel<<<dim3(BS), 256, 0, stream>>>(ctxb, merge_w, merge_b, ctx_out);
}
